// MoELayer_50697793962491
// MI455X (gfx1250) — compile-verified, matrix operand order checked by reading, not run on hardware
//
#include <hip/hip_runtime.h>
#include <hip/hip_bf16.h>
#include <math.h>

// MoE top-2 layer for MI455X (gfx1250, wave32, WMMA, async-LDS copies).
// Route first, compute only K=2 selected experts (137 GFLOP vs dense 550),
// bf16 WMMA w/ f32 accumulation; bf16 weight copies (128 MB) live in 192 MB L2.
// K-loop staging uses global_load_async_to_lds_b128 with double buffering.
// LDS tile rows padded to 40 elements (20 DWORDs) -> conflict-free ds_load_b128.

#define BB 2
#define SS 2048
#define DD 1024
#define HH 4096
#define EE 8
#define TT (BB * SS)     // 4096 tokens
#define NA (TT * 2)      // 8192 assignments (top-2)
#define LP 40            // padded LDS row stride (elements)

typedef __attribute__((ext_vector_type(16))) __bf16 v16bf;
typedef __attribute__((ext_vector_type(8)))  float  v8f;

union FragU { v16bf v; uint4 u[2]; };

__device__ __forceinline__ v8f vzero8() {
  v8f z;
#pragma unroll
  for (int i = 0; i < 8; ++i) z[i] = 0.0f;
  return z;
}

// 16-byte async copy global -> LDS (per-lane addresses; tracked by ASYNCcnt).
__device__ __forceinline__ void async_copy16(const __bf16* g, __bf16* lds) {
  unsigned laddr = (unsigned)(size_t)lds;          // flat addr[31:0] == LDS offset
  unsigned long long gaddr = (unsigned long long)(size_t)g;
  asm volatile("global_load_async_to_lds_b128 %0, %1, off"
               :: "v"(laddr), "v"(gaddr) : "memory");
}
__device__ __forceinline__ void wait_async_le4() {
  asm volatile("s_wait_asynccnt 0x4" ::: "memory");
}
__device__ __forceinline__ void wait_async_0() {
  asm volatile("s_wait_asynccnt 0x0" ::: "memory");
}

// ---------------------------------------------------------------- conv x -> bf16
__global__ void moe_conv_x(const float* __restrict__ x, __bf16* __restrict__ xb, int n) {
  for (int i = blockIdx.x * blockDim.x + threadIdx.x; i < n; i += gridDim.x * blockDim.x)
    xb[i] = (__bf16)x[i];
}

// ------------------------------------------- transpose+convert weights (per expert in z)
// src: R x C row-major f32 ; dst: C x R row-major bf16
__global__ void moe_transpose_bf16(const float* __restrict__ src, __bf16* __restrict__ dst,
                                   int R, int C) {
  __shared__ __bf16 tile[32][33];
  const float* s = src + (size_t)blockIdx.z * R * C;
  __bf16*      d = dst + (size_t)blockIdx.z * R * C;
  int c0 = blockIdx.x * 32, r0 = blockIdx.y * 32;
  for (int i = threadIdx.y; i < 32; i += 8)
    tile[i][threadIdx.x] = (__bf16)s[(size_t)(r0 + i) * C + (c0 + threadIdx.x)];
  __syncthreads();
  for (int i = threadIdx.y; i < 32; i += 8)
    d[(size_t)(c0 + i) * R + (r0 + threadIdx.x)] = tile[threadIdx.x][i];
}

// ---------------------------------------------------------------- router: top-2 softmax
__global__ void moe_router(const float* __restrict__ x, const float* __restrict__ wr,
                           const float* __restrict__ br,
                           int* __restrict__ eA, float* __restrict__ wA) {
  int wave = threadIdx.x >> 5, lane = threadIdx.x & 31;
  int t = blockIdx.x * 8 + wave;           // grid = T/8 blocks of 8 waves
  const float* xr = x + (size_t)t * DD;
  float acc[EE];
#pragma unroll
  for (int e = 0; e < EE; ++e) acc[e] = 0.0f;
  for (int d0 = lane; d0 < DD; d0 += 32) {
    float xv = xr[d0];
    const float* w = wr + (size_t)d0 * EE;
#pragma unroll
    for (int e = 0; e < EE; ++e) acc[e] += xv * w[e];
  }
#pragma unroll
  for (int off = 16; off; off >>= 1)
#pragma unroll
    for (int e = 0; e < EE; ++e) acc[e] += __shfl_xor(acc[e], off);
  if (lane == 0) {
    float l0 = -3.0e38f, l1 = -3.0e38f; int i0 = 0, i1 = 0;
#pragma unroll
    for (int e = 0; e < EE; ++e) {
      float l = acc[e] + br[e];
      if (l > l0) { l1 = l0; i1 = i0; l0 = l; i0 = e; }
      else if (l > l1) { l1 = l; i1 = e; }
    }
    // softmax -> top2 -> renormalize == softmax over the two top logits
    float w1 = expf(l1 - l0);
    float w0 = 1.0f / (1.0f + w1);
    w1 = 1.0f - w0;
    eA[t * 2 + 0] = i0; eA[t * 2 + 1] = i1;
    wA[t * 2 + 0] = w0; wA[t * 2 + 1] = w1;
  }
}

// ---------------------------------------------------------------- bucketing
__global__ void moe_init_counters(int* __restrict__ cnt) {
  if (threadIdx.x < 3 * EE) cnt[threadIdx.x] = 0;   // count | count2 | offset
}
__global__ void moe_count(const int* __restrict__ eA, int* __restrict__ cnt) {
  int i = blockIdx.x * blockDim.x + threadIdx.x;
  if (i < NA) atomicAdd(&cnt[eA[i]], 1);
}
__global__ void moe_offsets(const int* __restrict__ cnt, int* __restrict__ off) {
  if (threadIdx.x == 0) {
    int acc = 0;
    for (int e = 0; e < EE; ++e) { off[e] = acc; acc += cnt[e]; }
  }
}
__global__ void moe_bucket(const int* __restrict__ eA, const int* __restrict__ off,
                           int* __restrict__ cnt2, int* __restrict__ tok_by_slot,
                           int* __restrict__ slot_of) {
  int i = blockIdx.x * blockDim.x + threadIdx.x;
  if (i < NA) {
    int e = eA[i];
    int pos = atomicAdd(&cnt2[e], 1);
    int slot = off[e] + pos;
    tok_by_slot[slot] = i >> 1;   // token index
    slot_of[i] = slot;
  }
}

// ---------------------------------------------------------------- WMMA helpers
__device__ __forceinline__ v16bf load_frag_lds(const __bf16* rowp, int kb) {
  FragU f;
  f.u[0] = *(const uint4*)(rowp + kb);
  f.u[1] = *(const uint4*)(rowp + kb + 16);
  return f.v;
}

// ---------------------------------------------------------------- FFN1: h = gelu(Xg @ W1 + b1)
// grid (H/128, T/128, E), block 256 (8 waves, each 32x64 of a 128x128 tile)
__global__ void __launch_bounds__(256)
moe_ffn1(const __bf16* __restrict__ xb, const __bf16* __restrict__ w1t,
         const float* __restrict__ b1, const int* __restrict__ tok_by_slot,
         const int* __restrict__ cnt, const int* __restrict__ off,
         __bf16* __restrict__ hbuf) {
  const int e  = blockIdx.z;
  const int m0 = blockIdx.y * 128;
  const int n0 = blockIdx.x * 128;
  const int count = cnt[e];
  if (m0 >= count) return;
  const int base = off[e];

  __shared__ __align__(16) __bf16 As[2][128][LP];
  __shared__ __align__(16) __bf16 Bs[2][128][LP];

  const int tid = threadIdx.x;
  const int lane = tid & 31, wave = tid >> 5;
  const int wm = wave >> 1, wn = wave & 1;          // 4x2 wave grid
  const int rowS = tid >> 1;                        // staging: 2 threads/row
  const int colS = (tid & 1) * 16;                  // each thread moves 32 B/row

  int mr0 = m0 + rowS; if (mr0 > count - 1) mr0 = count - 1;
  const int tokA = tok_by_slot[base + mr0];
  const __bf16* aG = xb + (size_t)tokA * DD + colS;
  const __bf16* bG = w1t + ((size_t)e * HH + n0 + rowS) * DD + colS;

  v8f acc[2][4];
#pragma unroll
  for (int i = 0; i < 2; ++i)
#pragma unroll
    for (int j = 0; j < 4; ++j) acc[i][j] = vzero8();

  const int kb = (lane >> 4) * 8;
  const int rsel = lane & 15;

  // double-buffered async staging: 4 async b128 per wave per step
  {
    async_copy16(aG + 0, &As[0][rowS][colS]);
    async_copy16(aG + 8, &As[0][rowS][colS + 8]);
    async_copy16(bG + 0, &Bs[0][rowS][colS]);
    async_copy16(bG + 8, &Bs[0][rowS][colS + 8]);
  }
  for (int k0 = 0, s = 0; k0 < DD; k0 += 32, ++s) {
    const int cur = s & 1;
    if (k0 + 32 < DD) {
      const int nk = k0 + 32;
      async_copy16(aG + nk,     &As[cur ^ 1][rowS][colS]);
      async_copy16(aG + nk + 8, &As[cur ^ 1][rowS][colS + 8]);
      async_copy16(bG + nk,     &Bs[cur ^ 1][rowS][colS]);
      async_copy16(bG + nk + 8, &Bs[cur ^ 1][rowS][colS + 8]);
      wait_async_le4();          // oldest 4 (current tile) have landed
    } else {
      wait_async_0();
    }
    __syncthreads();

    v16bf af[2], bf[4];
#pragma unroll
    for (int im = 0; im < 2; ++im)
      af[im] = load_frag_lds(&As[cur][wm * 32 + im * 16 + rsel][0], kb);
#pragma unroll
    for (int in = 0; in < 4; ++in)
      bf[in] = load_frag_lds(&Bs[cur][wn * 64 + in * 16 + rsel][0], kb);
#pragma unroll
    for (int im = 0; im < 2; ++im)
#pragma unroll
      for (int in = 0; in < 4; ++in)
        acc[im][in] = __builtin_amdgcn_wmma_f32_16x16x32_bf16(
            false, af[im], false, bf[in], (short)0, acc[im][in], false, false);
    __syncthreads();             // readers done before buffer is overwritten
  }

  // epilogue: bias + exact GELU, store bf16 h
  const int rloc = (lane >> 4) * 8, cloc = lane & 15;
#pragma unroll
  for (int im = 0; im < 2; ++im)
#pragma unroll
    for (int in = 0; in < 4; ++in) {
      const int col = n0 + wn * 64 + in * 16 + cloc;
      const float bias = b1[(size_t)e * HH + col];
#pragma unroll
      for (int v = 0; v < 8; ++v) {
        const int m = wm * 32 + im * 16 + rloc + v;
        const int mr = m0 + m;
        if (mr < count) {
          float hv = acc[im][in][v] + bias;
          float g = 0.5f * hv * (1.0f + erff(hv * 0.70710678118654752f));
          hbuf[(size_t)(base + mr) * HH + col] = (__bf16)g;
        }
      }
    }
}

// ---------------------------------------------------------------- FFN2: y = h @ W2 + b2
// grid (D/128, T/128, E), block 256
__global__ void __launch_bounds__(256)
moe_ffn2(const __bf16* __restrict__ hbuf, const __bf16* __restrict__ w2t,
         const float* __restrict__ b2, const int* __restrict__ cnt,
         const int* __restrict__ off, float* __restrict__ ya) {
  const int e  = blockIdx.z;
  const int m0 = blockIdx.y * 128;
  const int n0 = blockIdx.x * 128;
  const int count = cnt[e];
  if (m0 >= count) return;
  const int base = off[e];

  __shared__ __align__(16) __bf16 As[2][128][LP];
  __shared__ __align__(16) __bf16 Bs[2][128][LP];

  const int tid = threadIdx.x;
  const int lane = tid & 31, wave = tid >> 5;
  const int wm = wave >> 1, wn = wave & 1;
  const int rowS = tid >> 1;
  const int colS = (tid & 1) * 16;

  int mr0 = m0 + rowS; if (mr0 > count - 1) mr0 = count - 1;
  const __bf16* aG = hbuf + (size_t)(base + mr0) * HH + colS;
  const __bf16* bG = w2t + ((size_t)e * DD + n0 + rowS) * HH + colS;

  v8f acc[2][4];
#pragma unroll
  for (int i = 0; i < 2; ++i)
#pragma unroll
    for (int j = 0; j < 4; ++j) acc[i][j] = vzero8();

  const int kb = (lane >> 4) * 8;
  const int rsel = lane & 15;

  {
    async_copy16(aG + 0, &As[0][rowS][colS]);
    async_copy16(aG + 8, &As[0][rowS][colS + 8]);
    async_copy16(bG + 0, &Bs[0][rowS][colS]);
    async_copy16(bG + 8, &Bs[0][rowS][colS + 8]);
  }
  for (int k0 = 0, s = 0; k0 < HH; k0 += 32, ++s) {
    const int cur = s & 1;
    if (k0 + 32 < HH) {
      const int nk = k0 + 32;
      async_copy16(aG + nk,     &As[cur ^ 1][rowS][colS]);
      async_copy16(aG + nk + 8, &As[cur ^ 1][rowS][colS + 8]);
      async_copy16(bG + nk,     &Bs[cur ^ 1][rowS][colS]);
      async_copy16(bG + nk + 8, &Bs[cur ^ 1][rowS][colS + 8]);
      wait_async_le4();
    } else {
      wait_async_0();
    }
    __syncthreads();

    v16bf af[2], bf[4];
#pragma unroll
    for (int im = 0; im < 2; ++im)
      af[im] = load_frag_lds(&As[cur][wm * 32 + im * 16 + rsel][0], kb);
#pragma unroll
    for (int in = 0; in < 4; ++in)
      bf[in] = load_frag_lds(&Bs[cur][wn * 64 + in * 16 + rsel][0], kb);
#pragma unroll
    for (int im = 0; im < 2; ++im)
#pragma unroll
      for (int in = 0; in < 4; ++in)
        acc[im][in] = __builtin_amdgcn_wmma_f32_16x16x32_bf16(
            false, af[im], false, bf[in], (short)0, acc[im][in], false, false);
    __syncthreads();
  }

  const int rloc = (lane >> 4) * 8, cloc = lane & 15;
#pragma unroll
  for (int im = 0; im < 2; ++im)
#pragma unroll
    for (int in = 0; in < 4; ++in) {
      const int col = n0 + wn * 64 + in * 16 + cloc;
      const float bias = b2[(size_t)e * DD + col];
#pragma unroll
      for (int v = 0; v < 8; ++v) {
        const int m = wm * 32 + im * 16 + rloc + v;
        const int mr = m0 + m;
        if (mr < count)
          ya[(size_t)(base + mr) * DD + col] = acc[im][in][v] + bias;
      }
    }
}

// ---------------------------------------------------------------- weighted combine
__global__ void moe_combine(const float* __restrict__ ya, const int* __restrict__ slot_of,
                            const float* __restrict__ wA, float* __restrict__ out, int n) {
  for (int i = blockIdx.x * blockDim.x + threadIdx.x; i < n; i += gridDim.x * blockDim.x) {
    int t = i >> 10;            // / DD
    int d = i & (DD - 1);
    int s0 = slot_of[t * 2 + 0], s1 = slot_of[t * 2 + 1];
    out[i] = wA[t * 2 + 0] * ya[(size_t)s0 * DD + d] +
             wA[t * 2 + 1] * ya[(size_t)s1 * DD + d];
  }
}

// ================================================================ launch
extern "C" void kernel_launch(void* const* d_in, const int* in_sizes, int n_in,
                              void* d_out, int out_size, void* d_ws, size_t ws_size,
                              hipStream_t stream) {
  const float* x  = (const float*)d_in[0];
  const float* wr = (const float*)d_in[1];
  const float* br = (const float*)d_in[2];
  const float* W1 = (const float*)d_in[3];
  const float* b1 = (const float*)d_in[4];
  const float* W2 = (const float*)d_in[5];
  const float* b2 = (const float*)d_in[6];
  float* out = (float*)d_out;

  // workspace carve (~232 MB)
  uintptr_t base = (uintptr_t)d_ws;
  size_t pos = 0;
  auto alloc = [&](size_t bytes) -> void* {
    void* p = (void*)(base + pos);
    pos += (bytes + 255) & ~(size_t)255;
    return p;
  };
  __bf16* xb   = (__bf16*)alloc((size_t)TT * DD * 2);        //   8 MB
  __bf16* w1t  = (__bf16*)alloc((size_t)EE * HH * DD * 2);   //  64 MB (H x D per expert)
  __bf16* w2t  = (__bf16*)alloc((size_t)EE * DD * HH * 2);   //  64 MB (D x H per expert)
  __bf16* hbuf = (__bf16*)alloc((size_t)NA * HH * 2);        //  64 MB
  float*  ya   = (float*) alloc((size_t)NA * DD * 4);        //  32 MB
  int*   eA    = (int*)  alloc((size_t)NA * 4);
  float* wA    = (float*)alloc((size_t)NA * 4);
  int* tok_by_slot = (int*)alloc((size_t)NA * 4);
  int* slot_of     = (int*)alloc((size_t)NA * 4);
  int* counters    = (int*)alloc(3 * EE * 4);   // count | count2 | offset
  int* cnt  = counters;
  int* cnt2 = counters + EE;
  int* offs = counters + 2 * EE;

  // 1) convert x to bf16
  moe_conv_x<<<2048, 256, 0, stream>>>(x, xb, TT * DD);
  // 2) transpose-convert weights: W1[e] (DxH) -> w1t[e] (HxD); W2[e] (HxD) -> w2t[e] (DxH)
  moe_transpose_bf16<<<dim3(HH / 32, DD / 32, EE), dim3(32, 8), 0, stream>>>(W1, w1t, DD, HH);
  moe_transpose_bf16<<<dim3(DD / 32, HH / 32, EE), dim3(32, 8), 0, stream>>>(W2, w2t, HH, DD);
  // 3) router
  moe_init_counters<<<1, 64, 0, stream>>>(counters);
  moe_router<<<TT / 8, 256, 0, stream>>>(x, wr, br, eA, wA);
  // 4) bucket tokens per expert
  moe_count<<<NA / 256, 256, 0, stream>>>(eA, cnt);
  moe_offsets<<<1, 32, 0, stream>>>(cnt, offs);
  moe_bucket<<<NA / 256, 256, 0, stream>>>(eA, offs, cnt2, tok_by_slot, slot_of);
  // 5) FFN passes (WMMA bf16, async-LDS double buffering)
  moe_ffn1<<<dim3(HH / 128, TT / 128, EE), 256, 0, stream>>>(xb, w1t, b1, tok_by_slot, cnt, offs, hbuf);
  moe_ffn2<<<dim3(DD / 128, TT / 128, EE), 256, 0, stream>>>(hbuf, w2t, b2, cnt, offs, ya);
  // 6) weighted combine into output
  moe_combine<<<2048, 256, 0, stream>>>(ya, slot_of, wA, out, TT * DD);

  (void)in_sizes; (void)n_in; (void)out_size; (void)ws_size;
}
